// MemoryBank_7267084665075
// MI455X (gfx1250) — matrix operand order repaired and checked by reading, not compiled, not run on hardware
//
#include <hip/hip_runtime.h>

// ---------------- problem constants (from reference setup_inputs) ----------
#define BATCH 64
#define KDIM  512      // "dim"  : GEMM K
#define TDIM  2048     // "T"    : rows (M direction)
#define SDIM  1024     // "size" : GEMM N, softmax axis
#define BT    32       // t-tile per workgroup (2 WMMA M-tiles per wave)
#define NMT   (BT / 16)        // 2 M-tiles
#define KCH   32       // K per bf16 WMMA
#define NKC   (KDIM / KCH)     // 16 chunks
#define NWAVE 8                // 256 threads = 8 wave32
#define SPW   (SDIM / NWAVE)   // 128 s-columns per wave
#define NST   (SPW / 16)       // 8 WMMA N-tiles per wave

#define TSTR  1028                            // padded tile row stride (floats)
#define A_STAGE_ELEMS (NKC * BT * 32)         // 16384 bf16 per hi/lo array
#define SMEM_BYTES    (BT * TSTR * 4 + BT * 4)  // tile + inv array

typedef __attribute__((ext_vector_type(16))) __bf16 v16bf;
typedef __attribute__((ext_vector_type(8)))  float  v8f;
typedef __attribute__((ext_vector_type(4)))  float  v4f;

static __device__ __forceinline__ v8f wmma_bf16(v16bf a, v16bf b, v8f c) {
  return __builtin_amdgcn_wmma_f32_16x16x32_bf16(
      false, a, false, b, (short)0, c, false, false);
}

// ---------------------------------------------------------------------------
// Prologue 1: m_sq[s] = sum_d units[d,s]^2
__global__ void msq_kernel(const float* __restrict__ units, float* __restrict__ msq) {
  int s = blockIdx.x * blockDim.x + threadIdx.x;
  if (s >= SDIM) return;
  float acc = 0.f;
  for (int d = 0; d < KDIM; ++d) {
    float u = units[(size_t)d * SDIM + s];
    acc += u * u;
  }
  msq[s] = acc;
}

// Prologue 2: split units into bf16 hi/lo, packed in WMMA B-fragment order:
// packed[(kc*SDIM + s)*32 + j] holds k = kc*32 + j for column s.
__global__ void pack_units_kernel(const float* __restrict__ units,
                                  __bf16* __restrict__ uhi,
                                  __bf16* __restrict__ ulo) {
  int idx = blockIdx.x * blockDim.x + threadIdx.x;   // d*SDIM + s (coalesced)
  if (idx >= KDIM * SDIM) return;
  int d = idx / SDIM;
  int s = idx % SDIM;
  float u  = units[idx];
  __bf16 h = (__bf16)u;
  __bf16 l = (__bf16)(u - (float)h);
  size_t o = ((size_t)(d >> 5) * SDIM + s) * 32 + (d & 31);
  uhi[o] = h;
  ulo[o] = l;
}

// ---------------------------------------------------------------------------
__global__ __launch_bounds__(256)
void bank_softmax_kernel(const float* __restrict__ H,
                         const __bf16* __restrict__ uhi,
                         const __bf16* __restrict__ ulo,
                         const float* __restrict__ msq,
                         float* __restrict__ out) {
  extern __shared__ char smem[];
  __bf16* aHi  = (__bf16*)smem;                        // 32 KB (phase 1-2)
  __bf16* aLo  = (__bf16*)(smem + 2 * A_STAGE_ELEMS);  // 32 KB (phase 1-2)
  float*  tile = (float*)smem;                         // ~128.5 KB (phase 3-5)
  float*  invArr = (float*)(smem + BT * TSTR * 4);     // BT floats

  const int tid  = threadIdx.x;
  const int wave = tid >> 5;
  const int lane = tid & 31;
  const int half = lane >> 4;
  const int mn   = lane & 15;
  const int b    = blockIdx.y;
  const int t0   = blockIdx.x * BT;
  const int s_base = wave * SPW;

  // ---- Phase 1: stage A tile (BT x KDIM of H) into LDS as bf16 hi/lo frags.
  // aHi[(kc*BT + m)*32 + half*16 + i], element i <-> k = kc*32+half*16+i
  {
    const float* Hb = H + (size_t)b * KDIM * TDIM + t0;
#pragma unroll 4
    for (int it = 0; it < (KDIM * BT) / 256; ++it) {
      int idx = tid + 256 * it;          // k*BT + m, m fastest -> coalesced
      int k = idx >> 5;
      int m = idx & 31;
      float v  = Hb[(size_t)k * TDIM + m];
      __bf16 h = (__bf16)v;
      __bf16 l = (__bf16)(v - (float)h);
      int ko  = k & 31;
      int pos = ((k >> 5) * BT + m) * 32 +
                ko;
      aHi[pos] = h;
      aLo[pos] = l;
    }
  }
  __syncthreads();

  // ---- Phase 2: split-bf16 GEMM; each B fragment feeds NMT WMMAs.
  v8f acc[NMT][NST] = {};

  for (int kc = 0; kc < NKC; ++kc) {
    v16bf ah[NMT], al[NMT];
#pragma unroll
    for (int mt = 0; mt < NMT; ++mt) {
      int o = (kc * BT + mt * 16 + mn) * 32 + half * 16;
      ah[mt] = *(const v16bf*)(aHi + o);
      al[mt] = *(const v16bf*)(aLo + o);
    }
    const __bf16* uhb = uhi + ((size_t)kc * SDIM) * 32 + half * 16;
    const __bf16* ulb = ulo + ((size_t)kc * SDIM) * 32 + half * 16;
#pragma unroll
    for (int st = 0; st < NST; ++st) {
      int s = s_base + st * 16 + mn;
      v16bf bh = *(const v16bf*)(uhb + (size_t)s * 32);
      v16bf bl = *(const v16bf*)(ulb + (size_t)s * 32);
#pragma unroll
      for (int mt = 0; mt < NMT; ++mt) {
        acc[mt][st] = wmma_bf16(al[mt], bh, acc[mt][st]);
        acc[mt][st] = wmma_bf16(ah[mt], bl, acc[mt][st]);
        acc[mt][st] = wmma_bf16(ah[mt], bh, acc[mt][st]);
      }
    }
  }
  __syncthreads();   // all waves done with aHi/aLo before tile reuse

  // ---- Phase 3: logits = 2*cross - msq[s] into tile[m][s]
  // C layout: VGPR r -> M = r + 8*half, N = mn
#pragma unroll
  for (int st = 0; st < NST; ++st) {
    int s = s_base + st * 16 + mn;
    float ms = msq[s];
#pragma unroll
    for (int mt = 0; mt < NMT; ++mt) {
#pragma unroll
      for (int r = 0; r < 8; ++r) {
        int m = mt * 16 + r + 8 * half;
        tile[m * TSTR + s] = 2.0f * acc[mt][st][r] - ms;
      }
    }
  }
  __syncthreads();

  // ---- Phase 4: softmax over s per row; 8 threads/row (contiguous 8-lane
  // group -> __shfl_xor width 8). Vectorized float4 scans.
  const int row = tid >> 3;          // 0..31
  const int col = tid & 7;
  float* rowp = tile + row * TSTR;

  float mx = -INFINITY;
  for (int j = 0; j < SDIM / 32; ++j) {
    v4f v = *(const v4f*)(rowp + col * 4 + 32 * j);
    mx = fmaxf(mx, fmaxf(fmaxf(v[0], v[1]), fmaxf(v[2], v[3])));
  }
#pragma unroll
  for (int off = 4; off > 0; off >>= 1) mx = fmaxf(mx, __shfl_xor(mx, off, 8));

  float sum = 0.f;
  for (int j = 0; j < SDIM / 32; ++j) {
    float* p = rowp + col * 4 + 32 * j;
    v4f v = *(const v4f*)p;
    v4f e;
#pragma unroll
    for (int i = 0; i < 4; ++i) e[i] = __expf(v[i] - mx);
    *(v4f*)p = e;
    sum += (e[0] + e[1]) + (e[2] + e[3]);
  }
#pragma unroll
  for (int off = 4; off > 0; off >>= 1) sum += __shfl_xor(sum, off, 8);
  if (col == 0) invArr[row] = 1.0f / sum;
  __syncthreads();

  // ---- Phase 5: transposed, fully-coalesced write-out. A wave writes
  // 4 s-rows x 32 t = 4 x 128B lines; non-temporal (streamed, never reread).
  float* outb = out + (size_t)b * SDIM * TDIM + t0;
  for (int it = 0; it < (BT * SDIM / 4) / 256; ++it) {   // 32 iters
    int flat = it * 256 + tid;
    int t4 = (flat & 7) * 4;       // 0,4,...,28
    int s  = flat >> 3;            // 0..1023
    v4f o;
#pragma unroll
    for (int i = 0; i < 4; ++i)
      o[i] = tile[(t4 + i) * TSTR + s] * invArr[t4 + i];
    __builtin_nontemporal_store(o, (v4f*)(outb + (size_t)s * TDIM + t4));
  }
}

// ---------------------------------------------------------------------------
extern "C" void kernel_launch(void* const* d_in, const int* in_sizes, int n_in,
                              void* d_out, int out_size, void* d_ws, size_t ws_size,
                              hipStream_t stream) {
  const float* H     = (const float*)d_in[0];   // (B, dim, T) f32
  const float* units = (const float*)d_in[1];   // (dim, size) f32
  float* out = (float*)d_out;                   // (B, size, T) f32

  // workspace layout: msq (4KB) | uhi (1MB) | ulo (1MB)
  float*  msq = (float*)d_ws;
  __bf16* uhi = (__bf16*)((char*)d_ws + 4096);
  __bf16* ulo = (__bf16*)((char*)d_ws + 4096 + (size_t)KDIM * SDIM * sizeof(__bf16));

  msq_kernel<<<SDIM / 256, 256, 0, stream>>>(units, msq);
  pack_units_kernel<<<(KDIM * SDIM) / 256, 256, 0, stream>>>(units, uhi, ulo);

  dim3 grid(TDIM / BT, BATCH);
  bank_softmax_kernel<<<grid, 256, SMEM_BYTES, stream>>>(H, uhi, ulo, msq, out);
}
